// BoxRLELoss_6502580486437
// MI455X (gfx1250) — matrix operand order repaired and checked by reading, not compiled, not run on hardware
//
#include <hip/hip_runtime.h>
#include <hip/hip_bf16.h>
#include <math.h>

// ---------------------------------------------------------------------------
// BoxRLELoss for gfx1250 (MI455X, wave32, WMMA).
// Phase 1: rotated-IoU geometry (scalar VALU, 1 thread/box)
// Phase 2: 6-layer RealNVP flow; 64x64 hidden GEMMs via v_wmma_f32_16x16x32_bf16
//          with weights pre-swizzled into WMMA B-fragment order and staged in LDS.
// Phase 3: Gaussian log-prob + masked-mean reduction (shfl + atomics).
// ---------------------------------------------------------------------------

#define NBOX_CONST 131072
#define HID 64

typedef __bf16 bf16_t;
typedef __attribute__((ext_vector_type(16))) __bf16 v16bf;
typedef __attribute__((ext_vector_type(8)))  __bf16 v8bf;
typedef __attribute__((ext_vector_type(8)))  float  v8f;

// ---- workspace layout (bytes) ----
#define OFF_ACC   0u            // 2 floats: [0]=loss sum, [1]=mask count
#define OFF_F     64u           // 12 * 392 floats (W1|b1|b2|W3|b3 per MLP)
#define F_FLOATS  (12*392)
#define F_BYTES   (F_FLOATS*4)                  // 18816
#define OFF_W2    (OFF_F + F_BYTES)             // 18880 (16B aligned)
#define W2_HALFS  (12*4096)
#define W2_BYTES  (W2_HALFS*2)                  // 98304
#define OFF_DIRX  117248u
#define OFF_DIRY  (OFF_DIRX + NBOX_CONST*4u)
#define OFF_BASE  (OFF_DIRY + NBOX_CONST*4u)
#define OFF_MASK  (OFF_BASE + NBOX_CONST*4u)

// ---- LDS layout for the flow kernel (bytes) ----
#define LDS_F_BYTES    F_BYTES                  // 18816
#define LDS_W2_BYTES   W2_BYTES                 // 98304
#define LDS_TILE_BYTES (8*16*64*2)              // 16384 (8 waves x 16x64 bf16)
#define LDS_Z_BYTES    (8*16*2*4)               // 1024
#define LDS_LD_BYTES   (8*16*4)                 // 512
#define LDS_TOTAL (LDS_F_BYTES + LDS_W2_BYTES + LDS_TILE_BYTES + LDS_Z_BYTES + LDS_LD_BYTES) // 135040

__device__ __forceinline__ unsigned short f2bu(float x) {
  __bf16 b = (__bf16)x; unsigned short u; __builtin_memcpy(&u, &b, 2); return u;
}
__device__ __forceinline__ float bu2f(unsigned short u) {
  __bf16 b; __builtin_memcpy(&b, &u, 2); return (float)b;
}

// ===========================================================================
__global__ void init_kernel(float* acc) {
  if (threadIdx.x < 2) acc[threadIdx.x] = 0.f;
}

// ===========================================================================
// Pack weights: F region (f32, small mats) + W2 in per-lane WMMA B-fragment
// order so the flow kernel's B fragments are two contiguous 16B LDS loads.
// blockIdx.x = p in [0,12): ist = p/6 (0=s,1=t), layer = p%6.
__global__ void pack_kernel(const float* __restrict__ sw1, const float* __restrict__ sb1,
                            const float* __restrict__ sw2, const float* __restrict__ sb2,
                            const float* __restrict__ sw3, const float* __restrict__ sb3,
                            const float* __restrict__ tw1, const float* __restrict__ tb1,
                            const float* __restrict__ tw2, const float* __restrict__ tb2,
                            const float* __restrict__ tw3, const float* __restrict__ tb3,
                            float* __restrict__ F, unsigned short* __restrict__ W2) {
  const int p = blockIdx.x;
  const int ist = p / 6, layer = p % 6;
  const float* w1 = ist ? tw1 : sw1;
  const float* b1 = ist ? tb1 : sb1;
  const float* w2 = ist ? tw2 : sw2;
  const float* b2 = ist ? tb2 : sb2;
  const float* w3 = ist ? tw3 : sw3;
  const float* b3 = ist ? tb3 : sb3;

  float* Fp = F + p * 392;
  for (int t = threadIdx.x; t < 392; t += blockDim.x) {
    float v = 0.f;
    if (t < 128)        v = w1[(layer*2 + (t>>6))*64 + (t & 63)];   // W1[k][c]
    else if (t < 192)   v = b1[layer*64 + (t-128)];                 // b1
    else if (t < 256)   v = b2[layer*64 + (t-192)];                 // b2
    else if (t < 384) { int k=(t-256)>>6, j=(t-256)&63;             // W3^T [k][j]
                        v = w3[(layer*64 + j)*2 + k]; }
    else if (t < 386)   v = b3[layer*2 + (t-384)];                  // b3
    Fp[t] = v;
  }

  unsigned short* Wp = W2 + p * 4096;
  for (int t = threadIdx.x; t < 4096; t += blockDim.x) {
    const int e    = t & 15;
    const int lane = (t >> 4) & 31;
    const int f    = t >> 9;          // fragment id = nt*2 + c
    const int nt   = f >> 1;
    const int c    = f & 1;
    const int K    = c*32 + (e < 8 ? (lane >> 4)*8 + e : 16 + (lane >> 4)*8 + (e - 8));      // B layout: lanes 0-15 K=0-7,16-23; 16-31 K=8-15,24-31
    const int Ncol = nt*16 + (lane & 15);
    Wp[t] = f2bu(w2[(layer*64 + K)*64 + Ncol]);
  }
}

// ===========================================================================
// Geometry: rotated IoU via Sutherland-Hodgman clipping. 1 thread / box.
__device__ __forceinline__ void rot_corners(const float b[5], float* cx, float* cy) {
  const float co = __cosf(b[4]), si = __sinf(b[4]);
  const float dx = 0.5f*b[2], dy = 0.5f*b[3];
  const float lx[4] = {-dx, dx, dx, -dx};
  const float ly[4] = {-dy, -dy, dy, dy};
  #pragma unroll
  for (int k = 0; k < 4; ++k) {
    cx[k] = b[0] + lx[k]*co - ly[k]*si;
    cy[k] = b[1] + lx[k]*si + ly[k]*co;
  }
}

__global__ void geom_kernel(const float* __restrict__ pred, const float* __restrict__ target,
                            const float* __restrict__ sigma, const float* __restrict__ tw,
                            float* __restrict__ dirx, float* __restrict__ diry,
                            float* __restrict__ base, float* __restrict__ maskv,
                            float* __restrict__ acc, int n) {
  const int i = blockIdx.x * blockDim.x + threadIdx.x;
  float m = 0.f;
  if (i < n) {
    float bp[5], bt[5];
    #pragma unroll
    for (int k = 0; k < 5; ++k) { bp[k] = pred[i*5+k]; bt[k] = target[i*5+k]; }
    float c1x[4], c1y[4], c2x[4], c2y[4];
    rot_corners(bp, c1x, c1y);
    rot_corners(bt, c2x, c2y);

    float px[16], py[16], qx[16], qy[16];
    int cnt = 4;
    #pragma unroll
    for (int k = 0; k < 4; ++k) { px[k] = c1x[k]; py[k] = c1y[k]; }

    for (int j = 0; j < 4 && cnt > 0; ++j) {
      const float ax = c2x[j], ay = c2y[j];
      const float bx = c2x[(j+1)&3], by = c2y[(j+1)&3];
      const float ddx = bx-ax, ddy = by-ay;
      int mm = 0;
      for (int k = 0; k < cnt; ++k) {
        const int k2 = (k+1 == cnt) ? 0 : k+1;
        const float sx = px[k],  sy = py[k];
        const float ex = px[k2], ey = py[k2];
        const float ss = ddx*(sy-ay) - ddy*(sx-ax);
        const float se = ddx*(ey-ay) - ddy*(ex-ax);
        const bool sin_ = ss >= 0.f, ein = se >= 0.f;
        if (sin_ != ein) {
          float den = ss - se;
          if (fabsf(den) < 1e-12f) den = 1.f;
          const float t = ss / den;
          if (mm < 16) { qx[mm] = sx + t*(ex-sx); qy[mm] = sy + t*(ey-sy); ++mm; }
        }
        if (ein && mm < 16) { qx[mm] = ex; qy[mm] = ey; ++mm; }
      }
      cnt = mm;
      for (int k = 0; k < cnt; ++k) { px[k] = qx[k]; py[k] = qy[k]; }
    }

    float area2 = 0.f;
    if (cnt >= 3) {
      for (int k = 0; k < cnt; ++k) {
        const int k2 = (k+1 == cnt) ? 0 : k+1;
        area2 += px[k]*py[k2] - px[k2]*py[k];
      }
    }
    const float inter = 0.5f * fabsf(area2);
    const float a1 = bp[2]*bp[3], a2 = bt[2]*bt[3];
    const float uni = fmaxf(a1 + a2 - inter, 1e-9f);
    const float iou = fmaxf(inter / uni, 1e-6f);

    float tws = 0.f;
    #pragma unroll
    for (int k = 0; k < 5; ++k) tws += tw[i*5+k];
    m = (tws * 0.2f > 0.f) ? 1.f : 0.f;

    const float sg  = 1.f / (1.f + expf(sigma[i]));   // 1 - sigmoid(x)
    const float err = (1.f - iou) / (sg + 1e-9f);
    const float dxc = bp[0]-bt[0], dyc = bp[1]-bt[1];
    const float nrm = sqrtf(dxc*dxc + dyc*dyc);
    dirx[i]  = err * dxc / nrm;
    diry[i]  = err * dyc / nrm;
    base[i]  = logf(sg) + logf(2.f*sg) + fabsf(err);
    maskv[i] = m;
  }
  // wave32 reduction of mask count, one atomic per wave
  #pragma unroll
  for (int off = 16; off > 0; off >>= 1) m += __shfl_xor(m, off, 32);
  if ((threadIdx.x & 31) == 0) atomicAdd(&acc[1], m);
}

// ===========================================================================
// One MLP (2->64->64->2) for a 16-sample tile held by one wave.
// Lane l owns (sample r = l&15, output component k = l>>4).
// The 64x64 hidden layer runs on v_wmma_f32_16x16x32_bf16.
__device__ __forceinline__ float mlp_eval(const float* __restrict__ F,
                                          const unsigned short* __restrict__ W2,
                                          unsigned short* __restrict__ tile,
                                          float z0_, float z1_, int lane) {
  const int r = lane & 15, half = lane >> 4;
  const int c0 = half * 32;

  // h1 = leaky_relu(z_ @ W1 + b1); lane writes 32 cols of its row (bf16 pairs)
  #pragma unroll
  for (int c = 0; c < 32; c += 2) {
    const int col = c0 + c;
    float ha = fmaf(z0_, F[col],   fmaf(z1_, F[64+col],   F[128+col]));
    float hb = fmaf(z0_, F[col+1], fmaf(z1_, F[64+col+1], F[128+col+1]));
    ha = ha > 0.f ? ha : 0.01f*ha;
    hb = hb > 0.f ? hb : 0.01f*hb;
    const unsigned int pk = (unsigned int)f2bu(ha) | ((unsigned int)f2bu(hb) << 16);
    *(unsigned int*)(tile + r*64 + col) = pk;
  }

  // A fragments (16x32 bf16): lane row r, K in [kb,kb+8) U [kb+16,kb+24), kb=half*8
  v16bf a0, a1;
  {
    const v8bf* row = (const v8bf*)(tile + r*64);   // 8 groups of 8 bf16
    const v8bf lo0 = row[half],     hi0 = row[half + 2];
    const v8bf lo1 = row[4 + half], hi1 = row[6 + half];
    #pragma unroll
    for (int e = 0; e < 8; ++e) {
      a0[e] = lo0[e]; a0[e+8] = hi0[e];
      a1[e] = lo1[e]; a1[e+8] = hi1[e];
    }
  }

  // h2 = leaky_relu(h1 @ W2 + b2): 4 N-tiles x 2 K-chunks of WMMA
  #pragma unroll
  for (int nt = 0; nt < 4; ++nt) {
    v8f accv;
    #pragma unroll
    for (int v = 0; v < 8; ++v) accv[v] = 0.f;
    #pragma unroll
    for (int c = 0; c < 2; ++c) {
      const unsigned short* bp = W2 + ((nt*2 + c)*32 + lane)*16;
      const v8bf blo = *(const v8bf*)bp;
      const v8bf bhi = *(const v8bf*)(bp + 8);
      v16bf bfrag;
      #pragma unroll
      for (int e = 0; e < 8; ++e) { bfrag[e] = blo[e]; bfrag[e+8] = bhi[e]; }
      const v16bf afrag = (c == 0) ? a0 : a1;
      accv = __builtin_amdgcn_wmma_f32_16x16x32_bf16(
          false, afrag, false, bfrag, (short)0, accv, false, false);
    }
    const int ncol = nt*16 + r;                 // D layout: lane holds col = lane&15
    const float bias = F[192 + ncol];
    #pragma unroll
    for (int v = 0; v < 8; ++v) {
      const int mrow = v + half*8;              // rows v (lanes<16) / v+8 (lanes>=16)
      float hv = accv[v] + bias;
      hv = hv > 0.f ? hv : 0.01f*hv;
      tile[mrow*64 + ncol] = f2bu(hv);
    }
  }

  // out_k = h2 @ W3 + b3 for this lane's component k = half
  const int k = half;
  float out = F[384 + k];
  const float* w3k = F + 256 + k*64;            // W3^T row, contiguous
  const unsigned short* trow = tile + r*64;
  #pragma unroll
  for (int g = 0; g < 8; ++g) {
    const v8bf hv = *(const v8bf*)(trow + g*8);
    #pragma unroll
    for (int e = 0; e < 8; ++e) out = fmaf((float)hv[e], w3k[g*8 + e], out);
  }
  return out;
}

// ===========================================================================
__global__ __launch_bounds__(256)
void flow_kernel(const float* __restrict__ dirx, const float* __restrict__ diry,
                 const float* __restrict__ base, const float* __restrict__ maskv,
                 const int* __restrict__ labels, const float* __restrict__ loc,
                 const float* __restrict__ cov,
                 const float* __restrict__ Fg, const unsigned short* __restrict__ W2g,
                 float* __restrict__ acc, int ntiles) {
  extern __shared__ char smem[];
  float*          Fl    = (float*)smem;
  unsigned short* W2l   = (unsigned short*)(smem + LDS_F_BYTES);
  unsigned short* tiles = (unsigned short*)(smem + LDS_F_BYTES + LDS_W2_BYTES);
  float*          zb    = (float*)(smem + LDS_F_BYTES + LDS_W2_BYTES + LDS_TILE_BYTES);
  float*          ldb   = zb + 8*32;

  // cooperative stage of all weights into LDS (resident for whole kernel)
  for (int t = threadIdx.x; t < F_BYTES/16; t += blockDim.x)
    ((float4*)Fl)[t] = ((const float4*)Fg)[t];
  for (int t = threadIdx.x; t < W2_BYTES/16; t += blockDim.x)
    ((uint4*)W2l)[t] = ((const uint4*)W2g)[t];
  __syncthreads();

  const int lane = threadIdx.x & 31;
  const int wid  = threadIdx.x >> 5;
  const int r    = lane & 15;
  const int half = lane >> 4;
  const int gw   = blockIdx.x * (blockDim.x >> 5) + wid;
  const int gstride = gridDim.x * (blockDim.x >> 5);

  unsigned short* tl = tiles + wid * 1024;      // 16x64 bf16 per wave
  float* z  = zb  + wid * 32;
  float* ld = ldb + wid * 16;

  for (int ti = gw; ti < ntiles; ti += gstride) {
    const int s0 = ti * 16;
    if (ti + gstride < ntiles) {                // prefetch next tile's inputs
      __builtin_prefetch(dirx + (ti + gstride)*16, 0, 1);
      __builtin_prefetch(diry + (ti + gstride)*16, 0, 1);
    }
    if (lane < 16) {
      z[lane*2 + 0] = dirx[s0 + lane];
      z[lane*2 + 1] = diry[s0 + lane];
      ld[lane] = 0.f;
    }

    for (int layer = 5; layer >= 0; --layer) {
      const int act = layer & 1;                // active output component
      const float zz0 = z[r*2], zz1 = z[r*2 + 1];
      const float z_0 = (act == 0) ? 0.f : zz0; // z_ = m * z
      const float z_1 = (act == 0) ? zz1 : 0.f;

      const float*          Fs  = Fl  + layer * 392;          // s-MLP (ist=0)
      const unsigned short* W2s = W2l + layer * 4096;
      const float*          Ft  = Fl  + (6 + layer) * 392;    // t-MLP (ist=1)
      const unsigned short* W2t = W2l + (6 + layer) * 4096;

      const float s_out = mlp_eval(Fs, W2s, tl, z_0, z_1, lane);
      const float t_out = mlp_eval(Ft, W2t, tl, z_0, z_1, lane);
      const float sv = tanhf(s_out);

      if (half == act) {                        // 16 lanes own the active comp
        const float zk = (act == 0) ? zz0 : zz1;
        z[r*2 + act] = (zk - t_out) * expf(-sv);
        ld[r] -= sv;
      }
    }

    float contrib = 0.f;
    if (lane < 16) {
      const int idx = s0 + lane;
      const float z0f = z[lane*2], z1f = z[lane*2 + 1];
      const int lab = labels[idx];
      const float mu0 = loc[lab*2], mu1 = loc[lab*2 + 1];
      const float ca = cov[lab*4+0], cb = cov[lab*4+1];
      const float cc = cov[lab*4+2], cd = cov[lab*4+3];
      const float det = ca*cd - cb*cc;
      const float dxm = z0f - mu0, dym = z1f - mu1;
      const float quad = (cd*dxm*dxm - (cb+cc)*dxm*dym + ca*dym*dym) / det;
      const float logp = -0.5f*quad - 0.5f*logf(det) - 1.8378770664093453f; // log(2*pi)
      contrib = (base[idx] - logp - ld[lane]) * maskv[idx];
    }
    #pragma unroll
    for (int off = 16; off > 0; off >>= 1) contrib += __shfl_xor(contrib, off, 32);
    if (lane == 0) atomicAdd(&acc[0], contrib);
  }
}

// ===========================================================================
__global__ void finalize_kernel(const float* __restrict__ acc, float* __restrict__ out) {
  const float cnt = acc[1];
  const float total = acc[0] / fmaxf(cnt, 1.f) * 0.1f;
  out[0] = (cnt > 0.f) ? total : 0.f;
}

// ===========================================================================
extern "C" void kernel_launch(void* const* d_in, const int* in_sizes, int n_in,
                              void* d_out, int out_size, void* d_ws, size_t ws_size,
                              hipStream_t stream) {
  const float* pred   = (const float*)d_in[0];
  const float* target = (const float*)d_in[1];
  const float* sigma  = (const float*)d_in[2];
  const int*   labels = (const int*)d_in[3];
  const float* tw     = (const float*)d_in[4];
  const float* loc    = (const float*)d_in[5];
  const float* cov    = (const float*)d_in[6];
  const float* sw1 = (const float*)d_in[7],  *sb1 = (const float*)d_in[8];
  const float* sw2 = (const float*)d_in[9],  *sb2 = (const float*)d_in[10];
  const float* sw3 = (const float*)d_in[11], *sb3 = (const float*)d_in[12];
  const float* tw1 = (const float*)d_in[13], *tb1 = (const float*)d_in[14];
  const float* tw2 = (const float*)d_in[15], *tb2 = (const float*)d_in[16];
  const float* tw3 = (const float*)d_in[17], *tb3 = (const float*)d_in[18];

  const int n = in_sizes[2];                    // N (sigma length)
  char* ws = (char*)d_ws;
  float*          acc   = (float*)(ws + OFF_ACC);
  float*          Fp    = (float*)(ws + OFF_F);
  unsigned short* W2p   = (unsigned short*)(ws + OFF_W2);
  float*          dirx  = (float*)(ws + OFF_DIRX);
  float*          diry  = (float*)(ws + OFF_DIRY);
  float*          baseb = (float*)(ws + OFF_BASE);
  float*          maskb = (float*)(ws + OFF_MASK);

  init_kernel<<<1, 32, 0, stream>>>(acc);
  pack_kernel<<<12, 256, 0, stream>>>(sw1, sb1, sw2, sb2, sw3, sb3,
                                      tw1, tb1, tw2, tb2, tw3, tb3, Fp, W2p);
  geom_kernel<<<(n + 255) / 256, 256, 0, stream>>>(pred, target, sigma, tw,
                                                   dirx, diry, baseb, maskb, acc, n);
  const int ntiles = n / 16;
  flow_kernel<<<512, 256, LDS_TOTAL, stream>>>(dirx, diry, baseb, maskb, labels, loc, cov,
                                               Fp, W2p, acc, ntiles);
  finalize_kernel<<<1, 1, 0, stream>>>(acc, (float*)d_out);
}
